// EncoderSRNN_3796751090323
// MI455X (gfx1250) — matrix operand order repaired and checked by reading, not compiled, not run on hardware
//
#include <hip/hip_runtime.h>
#include <hip/hip_bf16.h>
#include <math.h>

typedef __attribute__((ext_vector_type(16))) _Float16 v16h;
typedef __attribute__((ext_vector_type(8)))  _Float16 v8h;
typedef __attribute__((ext_vector_type(8)))  float    v8f;

#define VOC    32000
#define EDIM   256
#define HDIM   256
#define SSZ    128
#define SDIM   128
#define SDEPTH 2
#define NSEQ   64
#define BSZ    32
#define TSTEPS 127
#define NBUF (TSTEPS * BSZ * EDIM)   // 1,040,384
#define NSTK (BSZ * SSZ * SDIM)      // 524,288
#define NHID (BSZ * HDIM)            // 8,192

union Frag { v16h v; v8h h[2]; };

// A-matrix fragment (16x32 f16, row-major src, ld = K stride in halves)
// lanes 0-15: M=lane, K chunks [kb..kb+7] and [kb+16..kb+23]
// lanes 16-31: M=lane-16, K chunks [kb+8..kb+15] and [kb+24..kb+31]
__device__ __forceinline__ v16h load_frag_A(const _Float16* m, int ld, int mbase,
                                            int kbase, int lane) {
  int half = lane >> 4;
  const _Float16* p = m + (mbase + (lane & 15)) * ld + kbase + half * 8;
  Frag f;
  f.h[0] = *(const v8h*)(p);
  f.h[1] = *(const v8h*)(p + 16);
  return f.v;
}

// B-matrix fragment (32x16 f16). W is row-major [outdim][K]; B = W^T tile.
// lanes 0-15: N=lane, K chunks [kb..kb+7] and [kb+16..kb+23]; lanes 16-31: N=lane-16, K chunks [kb+8..kb+15] and [kb+24..kb+31]
__device__ __forceinline__ v16h load_frag_B(const _Float16* w, int ld, int nbase,
                                            int kbase, int lane) {
  int half = lane >> 4;
  const _Float16* p = w + (nbase + (lane & 15)) * ld + kbase + half * 8;
  Frag f;
  f.h[0] = *(const v8h*)(p);
  f.h[1] = *(const v8h*)(p + 16);
  return f.v;
}

// ---------------------------------------------------------------------------
// Init: f32->f16 weight conversion, buf0 = [emb gather ; zero pads],
// stack0 = broadcast(empty_elem), hid = 0.
// ---------------------------------------------------------------------------
__global__ __launch_bounds__(256)
void srnn_init(const int* __restrict__ inputs, const float* __restrict__ emb_W,
               const float* __restrict__ We2h, const float* __restrict__ Ws2h,
               const float* __restrict__ Wh2h, const float* __restrict__ Wh2s,
               const float* __restrict__ Ws2u, const float* __restrict__ emp,
               float* __restrict__ buf0, float* __restrict__ stack0,
               float* __restrict__ hid_ws, _Float16* __restrict__ w16) {
  const int stride = gridDim.x * 256;
  const int base = blockIdx.x * 256 + threadIdx.x;
  for (int i = base; i < NBUF; i += stride) {
    int tt = i >> 13;  // / (32*256)
    float v = 0.0f;
    if (tt < NSEQ) {
      int b = (i >> 8) & 31, e = i & 255;
      int tok = inputs[tt * BSZ + b];
      v = emb_W[(size_t)tok * EDIM + e];
    }
    buf0[i] = v;
  }
  for (int i = base; i < NSTK; i += stride) stack0[i] = emp[i & 127];
  for (int i = base; i < NHID; i += stride) hid_ws[i] = 0.0f;
  for (int i = base; i < 262144; i += stride) {
    float v;
    if      (i <  65536) v = We2h[i];
    else if (i < 131072) v = Ws2h[i -  65536];
    else if (i < 196608) v = Wh2h[i - 131072];
    else if (i < 229376) v = Wh2s[i - 196608];
    else                 v = Ws2u[i - 229376];
    w16[i] = (_Float16)v;
  }
}

// ---------------------------------------------------------------------------
// Per-step recurrent core: single workgroup, 8 wave32 waves, all GEMMs via
// v_wmma_f32_16x16x32_f16. ihid accumulators stay in registers between the
// ihid GEMMs and the hid-update GEMM.
// ---------------------------------------------------------------------------
__global__ __launch_bounds__(256)
void srnn_step(const float* __restrict__ buf_cur, const float* __restrict__ stack_cur,
               float* __restrict__ hid_ws, const _Float16* __restrict__ w16,
               const float* __restrict__ b_e2h, const float* __restrict__ b_s2h,
               const float* __restrict__ b_h2h, const float* __restrict__ W_h2i,
               const float* __restrict__ b_h2i, const float* __restrict__ b_h2s,
               const float* __restrict__ b_s2u, float* __restrict__ p_ws,
               float* __restrict__ pv_ws, float* __restrict__ uv_ws,
               float* __restrict__ out_outputs, float* __restrict__ out_hid,
               float* __restrict__ out_acts, int t) {
  __shared__ __attribute__((aligned(16))) _Float16 sX[BSZ * HDIM];    // inp -> hid -> hid_new
  __shared__ __attribute__((aligned(16))) _Float16 sTops[BSZ * HDIM]; // stack[:, :2, :]
  __shared__ __attribute__((aligned(16))) _Float16 sIh[BSZ * HDIM];   // ihid (f16)
  __shared__ float sInst[BSZ * 3];

  const int tid  = threadIdx.x;
  const int lane = tid & 31;
  const int wave = tid >> 5;
  const int half = lane >> 4;

  const _Float16* We2h = w16;
  const _Float16* Ws2h = w16 + 65536;
  const _Float16* Wh2h = w16 + 131072;
  const _Float16* Wh2s = w16 + 196608;
  const _Float16* Ws2u = w16 + 229376;

  // ---- Phase A: stage inp (buf[t]) and tops into LDS as f16
  const float* inp = buf_cur + (size_t)t * (BSZ * EDIM);
  for (int i = tid; i < BSZ * HDIM; i += 256) {
    sX[i] = (_Float16)inp[i];
    int b = i >> 8, j = i & 255;                  // tops: s in {0,1} flattened
    sTops[i] = (_Float16)stack_cur[b * (SSZ * SDIM) + j];
  }
  __syncthreads();

  // Tile ownership: 32x256 output = 2 m-tiles x 16 n-tiles; 4 tiles per wave.
  const int mt  = wave & 1;
  const int ntb = (wave >> 1) * 4;

  // ---- Phase B: ihid = inp@We2h^T + tops@Ws2h^T + biases (kept in regs)
  v8f cih[4];
  for (int ti = 0; ti < 4; ++ti) {
    const int nt = ntb + ti;
    v8f c = {0.f, 0.f, 0.f, 0.f, 0.f, 0.f, 0.f, 0.f};
    for (int kt = 0; kt < 8; ++kt) {
      v16h a  = load_frag_A(sX, HDIM, mt * 16, kt * 32, lane);
      v16h bm = load_frag_B(We2h, EDIM, nt * 16, kt * 32, lane);
      c = __builtin_amdgcn_wmma_f32_16x16x32_f16(false, a, false, bm, (short)0, c, false, false);
    }
    for (int kt = 0; kt < 8; ++kt) {
      v16h a  = load_frag_A(sTops, HDIM, mt * 16, kt * 32, lane);
      v16h bm = load_frag_B(Ws2h, HDIM, nt * 16, kt * 32, lane);
      c = __builtin_amdgcn_wmma_f32_16x16x32_f16(false, a, false, bm, (short)0, c, false, false);
    }
    const int n = nt * 16 + (lane & 15);
    const float bias = b_e2h[n] + b_s2h[n];
    for (int r = 0; r < 8; ++r) c[r] += bias;
    cih[ti] = c;
  }
  __syncthreads();  // all reads of sX(inp) done

  // ---- Phase C: spill ihid f16 (for the 3-wide head); reload hid into sX
  for (int ti = 0; ti < 4; ++ti) {
    const int nt = ntb + ti;
    const int n  = nt * 16 + (lane & 15);
    for (int r = 0; r < 8; ++r) {
      int m = mt * 16 + r + half * 8;
      sIh[m * HDIM + n] = (_Float16)cih[ti][r];
    }
  }
  for (int i = tid; i < BSZ * HDIM; i += 256) sX[i] = (_Float16)hid_ws[i];
  __syncthreads();

  // ---- Phase D: hid_new = relu(hid@Wh2h^T + b_h2h + ihid); inst = ihid@Wh2i^T + b
  v8f ch[4];
  for (int ti = 0; ti < 4; ++ti) {
    const int nt = ntb + ti;
    v8f c = cih[ti];
    for (int kt = 0; kt < 8; ++kt) {
      v16h a  = load_frag_A(sX, HDIM, mt * 16, kt * 32, lane);
      v16h bm = load_frag_B(Wh2h, HDIM, nt * 16, kt * 32, lane);
      c = __builtin_amdgcn_wmma_f32_16x16x32_f16(false, a, false, bm, (short)0, c, false, false);
    }
    const int n = nt * 16 + (lane & 15);
    const float bias = b_h2h[n];
    for (int r = 0; r < 8; ++r) c[r] = fmaxf(c[r] + bias, 0.0f);
    ch[ti] = c;
  }
  if (tid < BSZ * 3) {
    int b = tid / 3, j = tid % 3;
    float s = b_h2i[j];
    for (int k = 0; k < HDIM; ++k) s += (float)sIh[b * HDIM + k] * W_h2i[j * HDIM + k];
    sInst[tid] = s;
  }
  __syncthreads();

  // ---- Phase E: publish hid_new (LDS f16 + globals); softmax/pow gating
  for (int ti = 0; ti < 4; ++ti) {
    const int nt = ntb + ti;
    const int n  = nt * 16 + (lane & 15);
    for (int r = 0; r < 8; ++r) {
      int m = mt * 16 + r + half * 8;
      float v = ch[ti][r];
      sX[m * HDIM + n] = (_Float16)v;
      out_outputs[(size_t)t * NHID + m * HDIM + n] = v;
      out_hid[m * HDIM + n] = v;
      hid_ws[m * HDIM + n]  = v;
    }
  }
  if (tid < BSZ) {
    float i0 = sInst[tid * 3 + 0], i1 = sInst[tid * 3 + 1], i2 = sInst[tid * 3 + 2];
    float mx = fmaxf(i0, i1);
    float e0 = expf(i0 - mx), e1 = expf(i1 - mx);
    float a0 = e0 / (e0 + e1), a1 = e1 / (e0 + e1);
    float gamma = 1.0f + log1pf(expf(i2));
    a0 = powf(a0, gamma);
    a1 = powf(a1, gamma);
    float inv = 1.0f / (a0 + a1 + 1e-16f);
    a0 *= inv; a1 *= inv;
    p_ws[tid * 2 + 0] = a0;
    p_ws[tid * 2 + 1] = a1;
    out_acts[(size_t)t * (BSZ * 2) + tid * 2 + 0] = a0;
    out_acts[(size_t)t * (BSZ * 2) + tid * 2 + 1] = a1;
  }
  __syncthreads();

  // ---- Phase F: push_val = relu(hid_new@Wh2s^T + b); u_val = relu(tops@Ws2u^T + b)
  {
    const int nt0 = (wave >> 1) * 2;  // 32x128 = 2x8 tiles; 2 tiles/wave (x2 outputs)
    for (int ti = 0; ti < 2; ++ti) {
      const int nt = nt0 + ti;
      v8f c = {0.f, 0.f, 0.f, 0.f, 0.f, 0.f, 0.f, 0.f};
      v8f d = {0.f, 0.f, 0.f, 0.f, 0.f, 0.f, 0.f, 0.f};
      for (int kt = 0; kt < 8; ++kt) {
        v16h ah = load_frag_A(sX, HDIM, mt * 16, kt * 32, lane);
        v16h bh = load_frag_B(Wh2s, HDIM, nt * 16, kt * 32, lane);
        c = __builtin_amdgcn_wmma_f32_16x16x32_f16(false, ah, false, bh, (short)0, c, false, false);
        v16h at = load_frag_A(sTops, HDIM, mt * 16, kt * 32, lane);
        v16h bt = load_frag_B(Ws2u, HDIM, nt * 16, kt * 32, lane);
        d = __builtin_amdgcn_wmma_f32_16x16x32_f16(false, at, false, bt, (short)0, d, false, false);
      }
      const int n  = nt * 16 + (lane & 15);
      const float b1 = b_h2s[n], b2 = b_s2u[n];
      for (int r = 0; r < 8; ++r) {
        int m = mt * 16 + r + half * 8;
        pv_ws[m * SDIM + n] = fmaxf(c[r] + b1, 0.0f);
        uv_ws[m * SDIM + n] = fmaxf(d[r] + b2, 0.0f);
      }
    }
  }
}

// ---------------------------------------------------------------------------
// Wide elementwise phase: buf blend (1.04M elems) + stack push/pop blend
// (0.52M elems), double-buffered; all traffic stays in L2.
// ---------------------------------------------------------------------------
__global__ __launch_bounds__(256)
void srnn_update(const float* __restrict__ buf_cur, float* __restrict__ buf_nxt,
                 const float* __restrict__ stack_cur, float* __restrict__ stack_nxt,
                 const float* __restrict__ p_ws, const float* __restrict__ pv_ws,
                 const float* __restrict__ uv_ws, const float* __restrict__ emp,
                 float* __restrict__ out_top, float* __restrict__ out_stack,
                 int t, int is_last) {
  int idx = blockIdx.x * 256 + threadIdx.x;
  if (idx < NBUF) {
    int b = (idx >> 8) & 31;
    float pp = p_ws[b * 2 + 0], pq = p_ws[b * 2 + 1];
    float prev = (idx >= BSZ * EDIM) ? buf_cur[idx - BSZ * EDIM] : 0.0f;
    buf_nxt[idx] = pp * buf_cur[idx] + pq * prev;
  } else if (idx < NBUF + NSTK) {
    int i2 = idx - NBUF;
    int b = i2 >> 14;
    int s = (i2 >> 7) & 127;
    int d = i2 & 127;
    float pp = p_ws[b * 2 + 0], pq = p_ws[b * 2 + 1];
    float push = (s == 0) ? pv_ws[b * SDIM + d] : stack_cur[i2 - SDIM];
    float pop;
    if (s == 0)            pop = uv_ws[b * SDIM + d];
    else if (s < SSZ - 1)  pop = stack_cur[i2 + SDIM];  // up shifted: stack[b][s+1]
    else                   pop = emp[d];                // s == 127: empty refill
    float v = pp * push + pq * pop;
    stack_nxt[i2] = v;
    if (s == 0) out_top[(size_t)t * (BSZ * SDIM) + b * SDIM + d] = v;
    if (is_last) out_stack[i2] = v;
  }
}

extern "C" void kernel_launch(void* const* d_in, const int* in_sizes, int n_in,
                              void* d_out, int out_size, void* d_ws, size_t ws_size,
                              hipStream_t stream) {
  (void)in_sizes; (void)n_in; (void)out_size; (void)ws_size;
  const int*   inputs = (const int*)d_in[0];
  const float* emb_W  = (const float*)d_in[1];
  const float* W_e2h  = (const float*)d_in[2];
  const float* b_e2h  = (const float*)d_in[3];
  const float* W_s2h  = (const float*)d_in[4];
  const float* b_s2h  = (const float*)d_in[5];
  const float* W_h2h  = (const float*)d_in[6];
  const float* b_h2h  = (const float*)d_in[7];
  const float* W_h2i  = (const float*)d_in[8];
  const float* b_h2i  = (const float*)d_in[9];
  const float* W_h2s  = (const float*)d_in[10];
  const float* b_h2s  = (const float*)d_in[11];
  const float* W_s2u  = (const float*)d_in[12];
  const float* b_s2u  = (const float*)d_in[13];
  const float* emp    = (const float*)d_in[14];

  char* ws = (char*)d_ws;
  float*    buf[2] = { (float*)(ws),            (float*)(ws +  4161536) };
  float*    stk[2] = { (float*)(ws +  8323072), (float*)(ws + 10420224) };
  float*    hid_ws = (float*)(ws + 12517376);
  float*    pv_ws  = (float*)(ws + 12550144);
  float*    uv_ws  = (float*)(ws + 12566528);
  float*    p_ws   = (float*)(ws + 12582912);
  _Float16* w16    = (_Float16*)(ws + 12583168);

  float* out_outputs = (float*)d_out;           // [127][32][256]
  float* out_hid     = out_outputs + 1040384;   // [32][256]
  float* out_stack   = out_hid + 8192;          // [32][128][128]
  float* out_acts    = out_stack + 524288;      // [127][32][2]
  float* out_top     = out_acts + 8128;         // [127][32][128]

  srnn_init<<<2048, 256, 0, stream>>>(inputs, emb_W, W_e2h, W_s2h, W_h2h, W_h2s,
                                      W_s2u, emp, buf[0], stk[0], hid_ws, w16);

  const int upd_blocks = (NBUF + NSTK + 255) / 256;
  for (int t = 0; t < TSTEPS; ++t) {
    int cur = t & 1, nxt = cur ^ 1;
    srnn_step<<<1, 256, 0, stream>>>(buf[cur], stk[cur], hid_ws, w16, b_e2h, b_s2h,
                                     b_h2h, W_h2i, b_h2i, b_h2s, b_s2u, p_ws, pv_ws,
                                     uv_ws, out_outputs, out_hid, out_acts, t);
    srnn_update<<<upd_blocks, 256, 0, stream>>>(buf[cur], buf[nxt], stk[cur], stk[nxt],
                                                p_ws, pv_ws, uv_ws, emp, out_top,
                                                out_stack, t, (t == TSTEPS - 1) ? 1 : 0);
  }
}
